// SparseTriangleCrossAttention_37134287242031
// MI455X (gfx1250) — matrix operand order repaired and checked by reading, not compiled, not run on hardware
//
#include <hip/hip_runtime.h>
#include <hip/hip_bf16.h>

#define NN   4096
#define CS   384
#define CZ   256
#define HH   8
#define CG   16
#define NRBF 64
#define EN   65536
#define EEN  262144

typedef __bf16 bh16;
typedef __attribute__((ext_vector_type(16))) __bf16 v16bf;
typedef __attribute__((ext_vector_type(8)))  __bf16 v8bf;
typedef __attribute__((ext_vector_type(8)))  float  v8f;
typedef __attribute__((ext_vector_type(4)))  float  v4f;

static __device__ __forceinline__ v16bf cat16(v8bf lo, v8bf hi) {
    return __builtin_shufflevector(lo, hi, 0,1,2,3,4,5,6,7,8,9,10,11,12,13,14,15);
}
static __device__ __forceinline__ v8f wmma_bf16(v16bf a, v16bf b, v8f c) {
    return __builtin_amdgcn_wmma_f32_16x16x32_bf16(false, a, false, b, (short)0, c, false, false);
}
static __device__ __forceinline__ float sigmoidf_(float x) {
    return 1.0f / (1.0f + __expf(-x));
}
static __device__ __forceinline__ void atomicMaxFloat(float* addr, float val) {
    if (val >= 0.0f) atomicMax((int*)addr, __float_as_int(val));
    else             atomicMin((unsigned int*)addr, (unsigned int)__float_as_int(val));
}

// ---------------------------------------------------------------------------
// LayerNorm (256 cols) -> bf16 output.  One wave per row.
// ---------------------------------------------------------------------------
__global__ void ln_bf16_kernel(const float* __restrict__ x,
                               const float* __restrict__ g,
                               const float* __restrict__ b,
                               bh16* __restrict__ y, int rows) {
    int row  = blockIdx.x * (blockDim.x >> 5) + (threadIdx.x >> 5);
    int lane = threadIdx.x & 31;
    if (row >= rows) return;
    const float* xr = x + (size_t)row * CZ;
    float v[8], s = 0.f, s2 = 0.f;
#pragma unroll
    for (int i = 0; i < 8; ++i) { v[i] = xr[lane + i * 32]; s += v[i]; s2 += v[i] * v[i]; }
#pragma unroll
    for (int o = 16; o > 0; o >>= 1) { s += __shfl_xor(s, o, 32); s2 += __shfl_xor(s2, o, 32); }
    float mu = s * (1.f / 256.f);
    float var = s2 * (1.f / 256.f) - mu * mu;
    float rs = rsqrtf(var + 1e-5f);
    bh16* yr = y + (size_t)row * CZ;
#pragma unroll
    for (int i = 0; i < 8; ++i) {
        int c = lane + i * 32;
        yr[c] = (bh16)((v[i] - mu) * rs * g[c] + b[c]);
    }
}

// ---------------------------------------------------------------------------
// node_left / node_right projections: [N,384] x [384,16] + bias (tiny; VALU)
// ---------------------------------------------------------------------------
__global__ void node_proj_kernel(const float* __restrict__ nf,
                                 const float* __restrict__ Wl, const float* __restrict__ bl,
                                 const float* __restrict__ Wr, const float* __restrict__ br,
                                 float* __restrict__ outl, float* __restrict__ outr) {
    int idx = blockIdx.x * blockDim.x + threadIdx.x;
    if (idx >= NN * CG) return;
    int n = idx >> 4, c = idx & 15;
    const float* row = nf + (size_t)n * CS;
    float al = 0.f, ar = 0.f;
    for (int k = 0; k < CS; ++k) {
        float f = row[k];
        al = fmaf(f, Wl[k * CG + c], al);
        ar = fmaf(f, Wr[k * CG + c], ar);
    }
    outl[idx] = al + bl[c];
    outr[idx] = ar + br[c];
}

// ---------------------------------------------------------------------------
// Pack fp32 weight [K, ncols] into per-lane WMMA bf16 B-fragment order:
//   out[((nt*KT+kt)*32 + lane)*16 + j] = W[(kt*32 + (j<8 ? 8*half+j : 16+8*half+j-8))*ncols + nt*16+n]
// ---------------------------------------------------------------------------
__global__ void packB_kernel(const float* __restrict__ W, bh16* __restrict__ out,
                             int K, int ncols, int NoutPad) {
    int idx = blockIdx.x * blockDim.x + threadIdx.x;
    int KT = K >> 5;
    int total = (NoutPad >> 4) * KT * 512;
    if (idx >= total) return;
    int j    = idx & 15;
    int lane = (idx >> 4) & 31;
    int kt   = (idx >> 9) % KT;
    int nt   = (idx >> 9) / KT;
    int n = lane & 15, half = lane >> 4;
    int k   = kt * 32 + ((j < 8) ? (8 * half + j) : (16 + 8 * half + (j - 8)));
    int col = nt * 16 + n;
    float v = (col < ncols) ? W[(size_t)k * ncols + col] : 0.f;
    out[idx] = (bh16)v;
}

// ---------------------------------------------------------------------------
// Generic WMMA GEMM:  C[M,Nout] = act(A_bf16[M,K] @ Bpacked + bias)
// block = 256 threads (8 waves); wave w handles ntile = blockIdx.y*8 + w.
// Packed B is visited sequentially -> running pointer + prefetch 1 tile ahead.
// ---------------------------------------------------------------------------
__global__ void __launch_bounds__(256)
wmma_gemm_kernel(const bh16* __restrict__ A, const bh16* __restrict__ Bp,
                 const float* __restrict__ bias, float* __restrict__ C,
                 int K, int Nout, int act_sigmoid) {
    const int lane = threadIdx.x & 31;
    const int wave = threadIdx.x >> 5;
    const int mt   = blockIdx.x;
    const int nt   = blockIdx.y * 8 + wave;
    const int half = lane >> 4, n = lane & 15;
    const int KT   = K >> 5;
    const int m    = mt * 16 + n;
    const v8bf* ap = (const v8bf*)(A + (size_t)m * K) + half;
    const v8bf* bb = (const v8bf*)Bp + ((size_t)nt * KT * 64 + (size_t)lane * 2);
    v8f acc = {};
    for (int kt = 0; kt < KT; ++kt) {
        __builtin_prefetch((const void*)(bb + 64), 0, 0);   // next B tile
        v8bf alo = ap[0];
        v8bf ahi = ap[2];
        v8bf blo = bb[0];
        v8bf bhi = bb[1];
        acc = wmma_bf16(cat16(alo, ahi), cat16(blo, bhi), acc);
        ap += 4;
        bb += 64;
    }
    int gcol = nt * 16 + n;
    float bv = bias ? bias[gcol] : 0.f;
#pragma unroll
    for (int r = 0; r < 8; ++r) {
        int row = mt * 16 + r + half * 8;
        float v = acc[r] + bv;
        if (act_sigmoid) v = sigmoidf_(v);
        C[(size_t)row * Nout + gcol] = v;
    }
}

// ---------------------------------------------------------------------------
// Triangle bias kernel: per group of 16 edge-edge pairs (one wave each):
//   t = sigmoid(outer(nl,nr) @ Wbg + bbg) * (rbf(d) @ Wdb + bdb)   [16 x 256]
//   bias = t @ Wtb                                                  [16 x 8]
// outer/rbf A-fragments built in registers; t staged to LDS (bf16) and
// re-read as A-fragments for the final WMMA.  All packed-B operands are
// visited strictly sequentially across the (nt,kt) loops -> streaming
// pointers with one-tile-ahead prefetch.
// ---------------------------------------------------------------------------
#define TB_WAVES 4
__global__ void __launch_bounds__(128)
triangle_bias_kernel(const int* __restrict__ eei,      // [2, EE] (row0=dst, row1=src)
                     const int* __restrict__ sei,      // [2, E]
                     const int* __restrict__ dei,      // [2, E]
                     const float* __restrict__ node_left,   // [N,16]
                     const float* __restrict__ node_right,  // [N,16]
                     const float* __restrict__ trans,       // [N,3]
                     const bh16* __restrict__ Wbg_p,   // packed KT=8, NT=16
                     const float* __restrict__ bbg,
                     const bh16* __restrict__ Wdb_p,   // packed KT=2, NT=16
                     const float* __restrict__ bdb,
                     const bh16* __restrict__ Wtb_p,   // packed KT=8, NT=1 (cols 8..15 zero)
                     float* __restrict__ bias_out)     // [EE, 8]
{
    __shared__ __align__(16) bh16 tshm[TB_WAVES][16 * CZ];
    const int lane = threadIdx.x & 31;
    const int wave = threadIdx.x >> 5;
    const int grp  = blockIdx.x * TB_WAVES + wave;
    const int mrow = lane & 15, half = lane >> 4;
    const int ee   = grp * 16 + mrow;

    const int eed = eei[ee];        // dst edge (key/value side)
    const int ees = eei[EEN + ee];  // src edge (query side)
    const int i1  = sei[ees];       // src_edge_index[0, ees]
    const int i2  = dei[eed];       // dst_edge_index[0, eed]

    float nl[16];
#pragma unroll
    for (int k = 0; k < 16; ++k) nl[k] = node_left[(size_t)i1 * 16 + k];
    float nr[8];
#pragma unroll
    for (int k = 0; k < 8; ++k) nr[k] = node_right[(size_t)i2 * 16 + half * 8 + k];

    float dx = trans[i1 * 3 + 0] - trans[i2 * 3 + 0] + 1e-8f;
    float dy = trans[i1 * 3 + 1] - trans[i2 * 3 + 1] + 1e-8f;
    float dz = trans[i1 * 3 + 2] - trans[i2 * 3 + 2] + 1e-8f;
    float d  = sqrtf(dx * dx + dy * dy + dz * dz);

    // outer-product A fragments (K = 256, 8 ktiles)
    v16bf afrag[8];
#pragma unroll
    for (int kt = 0; kt < 8; ++kt) {
        v16bf f;
        float a0 = nl[2 * kt], a1 = nl[2 * kt + 1];
#pragma unroll
        for (int j = 0; j < 8; ++j) {
            f[j]     = (bh16)(a0 * nr[j]);
            f[j + 8] = (bh16)(a1 * nr[j]);
        }
        afrag[kt] = f;
    }
    // RBF A fragments (K = 64, 2 ktiles); mu_r = 20*r/63, sigma = 20/64
    v16bf rfrag[2];
#pragma unroll
    for (int kt = 0; kt < 2; ++kt) {
        v16bf f;
#pragma unroll
        for (int j = 0; j < 16; ++j) {
            int K = kt * 32 + ((j < 8) ? (8 * half + j) : (16 + 8 * half + (j - 8)));
            float mu = 20.f * (float)K * (1.f / 63.f);
            float z  = (d - mu) * 3.2f;
            f[j] = (bh16)__expf(-z * z);
        }
        rfrag[kt] = f;
    }

    const int n = lane & 15;
    // streaming packed-B pointers (strictly sequential over the nt/kt loops)
    const v8bf* bgp = (const v8bf*)Wbg_p + (size_t)lane * 2;
    const v8bf* bdp = (const v8bf*)Wdb_p + (size_t)lane * 2;
    for (int nt = 0; nt < 16; ++nt) {
        __builtin_prefetch((const void*)(bgp + 512), 0, 0);  // next nt's gate tiles
        v8f accg = {};
#pragma unroll
        for (int kt = 0; kt < 8; ++kt) {
            accg = wmma_bf16(afrag[kt], cat16(bgp[0], bgp[1]), accg);
            bgp += 64;
        }
        v8f accd = {};
#pragma unroll
        for (int kt = 0; kt < 2; ++kt) {
            accd = wmma_bf16(rfrag[kt], cat16(bdp[0], bdp[1]), accd);
            bdp += 64;
        }
        int col = nt * 16 + n;
        float bg = bbg[col], bd = bdb[col];
#pragma unroll
        for (int r = 0; r < 8; ++r) {
            float t = sigmoidf_(accg[r] + bg) * (accd[r] + bd);
            tshm[wave][(r + 8 * half) * CZ + col] = (bh16)t;
        }
    }
    __syncthreads();

    // bias = t @ Wtb   (A from LDS, row-major bf16)
    v8f accb = {};
    const bh16* trow = &tshm[wave][mrow * CZ];
    const v8bf* tbp  = (const v8bf*)Wtb_p + (size_t)lane * 2;
#pragma unroll
    for (int kt = 0; kt < 8; ++kt) {
        v8bf alo = *((const v8bf*)trow + kt * 4 + half);
        v8bf ahi = *((const v8bf*)trow + kt * 4 + 2 + half);
        accb = wmma_bf16(cat16(alo, ahi), cat16(tbp[0], tbp[1]), accb);
        tbp += 64;
    }
    if (n < 8) {
#pragma unroll
        for (int r = 0; r < 8; ++r)
            bias_out[(size_t)(grp * 16 + r + 8 * half) * HH + n] = accb[r];
    }
}

// ---------------------------------------------------------------------------
// Softmax pipeline (irregular gather; atomics)
// ---------------------------------------------------------------------------
__global__ void init_kernel(float* __restrict__ segmax, float* __restrict__ denom,
                            float* __restrict__ upd) {
    int idx = blockIdx.x * blockDim.x + threadIdx.x;
    if (idx < EN * CZ) upd[idx] = 0.f;
    if (idx < EN * HH) { segmax[idx] = -3.0e38f; denom[idx] = 0.f; }
}

__global__ void attn_logits_kernel(const int* __restrict__ eei,
                                   const float* __restrict__ q,
                                   const float* __restrict__ kv,
                                   const float* __restrict__ bias,
                                   float* __restrict__ attn,
                                   float* __restrict__ segmax) {
    int idx = blockIdx.x * blockDim.x + threadIdx.x;
    if (idx >= EEN * HH) return;
    int ee = idx >> 3, h = idx & 7;
    int eed = eei[ee], ees = eei[EEN + ee];
    const v4f* qp = (const v4f*)(q + (size_t)ees * CZ + h * 32);
    const v4f* kp = (const v4f*)(kv + (size_t)eed * (2 * CZ) + h * 32);
    float s = 0.f;
#pragma unroll
    for (int i = 0; i < 8; ++i) {
        v4f a = qp[i], b = kp[i];
        s = fmaf(a[0], b[0], s); s = fmaf(a[1], b[1], s);
        s = fmaf(a[2], b[2], s); s = fmaf(a[3], b[3], s);
    }
    float a = s * 0.0625f + bias[idx];   // 1/sqrt(256)
    attn[idx] = a;
    atomicMaxFloat(&segmax[ees * HH + h], a);
}

__global__ void attn_exp_kernel(const int* __restrict__ eei,
                                float* __restrict__ attn,
                                const float* __restrict__ segmax,
                                float* __restrict__ denom) {
    int idx = blockIdx.x * blockDim.x + threadIdx.x;
    if (idx >= EEN * HH) return;
    int ee = idx >> 3, h = idx & 7;
    int ees = eei[EEN + ee];
    float ex = __expf(attn[idx] - segmax[ees * HH + h]);
    attn[idx] = ex;
    atomicAdd(&denom[ees * HH + h], ex);
}

__global__ void attn_scatter_kernel(const int* __restrict__ eei,
                                    const float* __restrict__ attn,
                                    const float* __restrict__ denom,
                                    const float* __restrict__ kv,
                                    float* __restrict__ upd) {
    int idx = blockIdx.x * blockDim.x + threadIdx.x;
    if (idx >= EEN * HH) return;
    int ee = idx >> 3, h = idx & 7;
    int eed = eei[ee], ees = eei[EEN + ee];
    float w = attn[idx] / (denom[ees * HH + h] + 1e-16f);
    const float* vp = kv + (size_t)eed * (2 * CZ) + CZ + h * 32;
    float* up = upd + (size_t)ees * CZ + h * 32;
#pragma unroll
    for (int d = 0; d < 32; ++d) atomicAdd(&up[d], vp[d] * w);
}

__global__ void gate_mul_kernel(const float* __restrict__ upd,
                                const float* __restrict__ gate,
                                bh16* __restrict__ ug) {
    int idx = blockIdx.x * blockDim.x + threadIdx.x;
    if (idx >= EN * CZ) return;
    ug[idx] = (bh16)(upd[idx] * gate[idx]);
}

// ---------------------------------------------------------------------------
// Launch
// ---------------------------------------------------------------------------
static inline unsigned cdiv(size_t a, unsigned b) { return (unsigned)((a + b - 1) / b); }

extern "C" void kernel_launch(void* const* d_in, const int* in_sizes, int n_in,
                              void* d_out, int out_size, void* d_ws, size_t ws_size,
                              hipStream_t stream) {
    (void)in_sizes; (void)n_in; (void)out_size; (void)ws_size;
    const float* node_features = (const float*)d_in[0];
    const float* node_trans    = (const float*)d_in[1];
    const float* dst_ef = (const float*)d_in[2];
    const float* src_ef = (const float*)d_in[3];
    const int*   dei    = (const int*)d_in[4];
    const int*   sei    = (const int*)d_in[5];
    const int*   eei    = (const int*)d_in[6];
    const float *Wnl = (const float*)d_in[7],  *bnl = (const float*)d_in[8];
    const float *Wnr = (const float*)d_in[9],  *bnr = (const float*)d_in[10];
    const float *Wbg = (const float*)d_in[11], *bbg = (const float*)d_in[12];
    const float *Wdb = (const float*)d_in[13], *bdb = (const float*)d_in[14];
    const float *Wtb = (const float*)d_in[15];
    const float *lndg = (const float*)d_in[16], *lndb = (const float*)d_in[17];
    const float *lnsg = (const float*)d_in[18], *lnsb = (const float*)d_in[19];
    const float *Wq  = (const float*)d_in[20], *bq  = (const float*)d_in[21];
    const float *Wkv = (const float*)d_in[22], *bkv = (const float*)d_in[23];
    const float *Wout= (const float*)d_in[24], *bout= (const float*)d_in[25];
    const float *Wgt = (const float*)d_in[26], *bgt = (const float*)d_in[27];

    char* ws = (char*)d_ws;
    size_t off = 0;
#define WS_ALLOC(type, name, count) \
    type* name = (type*)(ws + off); off += (((size_t)(count) * sizeof(type)) + 255) & ~(size_t)255;
    WS_ALLOC(bh16,  srcf,   (size_t)EN * CZ)       // LN(src) bf16
    WS_ALLOC(bh16,  dstf,   (size_t)EN * CZ)       // LN(dst) bf16
    WS_ALLOC(float, nleft,  (size_t)NN * CG)
    WS_ALLOC(float, nright, (size_t)NN * CG)
    WS_ALLOC(bh16,  Wq_p,   (size_t)8 * 16 * 512)
    WS_ALLOC(bh16,  Wkv_p,  (size_t)8 * 32 * 512)
    WS_ALLOC(bh16,  Wgt_p,  (size_t)8 * 16 * 512)
    WS_ALLOC(bh16,  Wout_p, (size_t)8 * 16 * 512)
    WS_ALLOC(bh16,  Wbg_p,  (size_t)8 * 16 * 512)
    WS_ALLOC(bh16,  Wdb_p,  (size_t)2 * 16 * 512)
    WS_ALLOC(bh16,  Wtb_p,  (size_t)8 * 1 * 512)
    WS_ALLOC(float, qbuf,   (size_t)EN * CZ)
    WS_ALLOC(float, kvbuf,  (size_t)EN * 2 * CZ)
    WS_ALLOC(float, gatebuf,(size_t)EN * CZ)
    WS_ALLOC(float, biasEE, (size_t)EEN * HH)
    WS_ALLOC(float, attn,   (size_t)EEN * HH)
    WS_ALLOC(float, segmax, (size_t)EN * HH)
    WS_ALLOC(float, denom,  (size_t)EN * HH)
    WS_ALLOC(float, upd,    (size_t)EN * CZ)
    WS_ALLOC(bh16,  ug,     (size_t)EN * CZ)
#undef WS_ALLOC

    const unsigned B = 256;
    // 1. pack weights into WMMA B-fragment layout
    packB_kernel<<<cdiv((size_t)8 * 16 * 512, B), B, 0, stream>>>(Wq,  Wq_p,  256, 256, 256);
    packB_kernel<<<cdiv((size_t)8 * 32 * 512, B), B, 0, stream>>>(Wkv, Wkv_p, 256, 512, 512);
    packB_kernel<<<cdiv((size_t)8 * 16 * 512, B), B, 0, stream>>>(Wgt, Wgt_p, 256, 256, 256);
    packB_kernel<<<cdiv((size_t)8 * 16 * 512, B), B, 0, stream>>>(Wout,Wout_p,256, 256, 256);
    packB_kernel<<<cdiv((size_t)8 * 16 * 512, B), B, 0, stream>>>(Wbg, Wbg_p, 256, 256, 256);
    packB_kernel<<<cdiv((size_t)2 * 16 * 512, B), B, 0, stream>>>(Wdb, Wdb_p,  64, 256, 256);
    packB_kernel<<<cdiv((size_t)8 * 1 * 512,  B), B, 0, stream>>>(Wtb, Wtb_p, 256,   8,  16);

    // 2. layernorms -> bf16
    ln_bf16_kernel<<<cdiv(EN, 8), B, 0, stream>>>(dst_ef, lndg, lndb, dstf, EN);
    ln_bf16_kernel<<<cdiv(EN, 8), B, 0, stream>>>(src_ef, lnsg, lnsb, srcf, EN);

    // 3. node projections
    node_proj_kernel<<<cdiv((size_t)NN * CG, B), B, 0, stream>>>(
        node_features, Wnl, bnl, Wnr, bnr, nleft, nright);

    // 4. q / kv / gate projections (WMMA)
    wmma_gemm_kernel<<<dim3(EN / 16, 2), B, 0, stream>>>(srcf, Wq_p,  bq,  qbuf,   256, 256, 0);
    wmma_gemm_kernel<<<dim3(EN / 16, 4), B, 0, stream>>>(dstf, Wkv_p, bkv, kvbuf,  256, 512, 0);
    wmma_gemm_kernel<<<dim3(EN / 16, 2), B, 0, stream>>>(srcf, Wgt_p, bgt, gatebuf,256, 256, 1);

    // 5. triangle bias (dominant WMMA kernel)
    triangle_bias_kernel<<<(EEN / 16) / TB_WAVES, 32 * TB_WAVES, 0, stream>>>(
        eei, sei, dei, nleft, nright, node_trans,
        Wbg_p, bbg, Wdb_p, bdb, Wtb_p, biasEE);

    // 6. segment softmax + weighted scatter
    init_kernel<<<cdiv((size_t)EN * CZ, B), B, 0, stream>>>(segmax, denom, upd);
    attn_logits_kernel<<<cdiv((size_t)EEN * HH, B), B, 0, stream>>>(eei, qbuf, kvbuf, biasEE, attn, segmax);
    attn_exp_kernel<<<cdiv((size_t)EEN * HH, B), B, 0, stream>>>(eei, attn, segmax, denom);
    attn_scatter_kernel<<<cdiv((size_t)EEN * HH, B), B, 0, stream>>>(eei, attn, denom, kvbuf, upd);

    // 7. gating + output projection (WMMA)
    gate_mul_kernel<<<cdiv((size_t)EN * CZ, B), B, 0, stream>>>(upd, gatebuf, ug);
    wmma_gemm_kernel<<<dim3(EN / 16, 2), B, 0, stream>>>(ug, Wout_p, bout, (float*)d_out, 256, 256, 0);
}
